// ClassicalSelfAttention_65481071408671
// MI455X (gfx1250) — compile-verified, matrix operand order checked by reading, not run on hardware
//
#include <hip/hip_runtime.h>

// ---------------------------------------------------------------------------
// ClassicalSelfAttention on MI455X (gfx1250, wave32, WMMA bf16)
//   q = (X @ Wq^T) @ R ; k = (X @ Wk^T) @ Ent ; v = X @ Wv^T
//   per-head (H=8, D=64) softmax(q k^T / 8) v  -> [B,S,E] fp32
// All matmuls via v_wmma_f32_16x16x32_bf16 (fp32 accumulate).
// Round 3: fix TDM builtin arity (6-arg amdgpu-toolchain variant).
// ---------------------------------------------------------------------------

#define EMBED 512
#define SEQ   4096
#define BATCH 2
#define HEADS 8
#define HDIM  64

typedef __attribute__((ext_vector_type(16))) __bf16 v16bf;
typedef __attribute__((ext_vector_type(8)))  __bf16 v8bf;
typedef __attribute__((ext_vector_type(8)))  float  v8f;

// TDM descriptor vector types (6-arg builtin: clang-23 / therock-10.0)
typedef __attribute__((ext_vector_type(4))) unsigned int tdm_v4u;
typedef __attribute__((ext_vector_type(8))) int          tdm_v8i;
typedef __attribute__((ext_vector_type(4))) int          tdm_v4i;

#if defined(__has_builtin)
#if __has_builtin(__builtin_amdgcn_tensor_load_to_lds)
#define HAVE_TDM 1
#endif
#endif

static __device__ __forceinline__ __bf16 f2bf(float f) {
  union { float f; unsigned u; } in; in.f = f;
  unsigned r = in.u + 0x7FFFu + ((in.u >> 16) & 1u);   // round-to-nearest-even
  union { unsigned short u; __bf16 b; } out;
  out.u = (unsigned short)(r >> 16);
  return out.b;
}

// Load one 16x32 (A) or 32x16 (Bt) bf16 WMMA operand fragment.
// Caller passes p = row_base + kbase where kbase = (lane>>4)*8.
// Lanes 0-15: elems 0..7 = K+0..7, elems 8..15 = K+16..23 (lanes 16-31: +8).
static __device__ __forceinline__ v16bf load_frag(const __bf16* p) {
  v8bf lo = *(const v8bf*)(p);
  v8bf hi = *(const v8bf*)(p + 16);
  return __builtin_shufflevector(lo, hi, 0,1,2,3,4,5,6,7,8,9,10,11,12,13,14,15);
}

static __device__ __forceinline__ v8f wmma_bf16(v16bf a, v16bf b, v8f c) {
  return __builtin_amdgcn_wmma_f32_16x16x32_bf16(
      /*neg_a=*/false, a, /*neg_b=*/false, b,
      /*c_mod=*/(short)0, c, /*reuse_a=*/false, /*reuse_b=*/false);
}

#ifdef HAVE_TDM
// Issue a TDM load of a 64x64 bf16 tile (row stride EMBED) into LDS.
// Descriptor layout per CDNA5 ISA (D# groups 0/1).
// 6-arg builtin signature: (v4u g0, v8i g1, v4i g2, v4i g3, v8i, i32 cpol).
static __device__ __forceinline__ void tdm_prefetch_tile(const __bf16* gptr,
                                                         unsigned lds_off,
                                                         int rows_left) {
  unsigned long long ga = (unsigned long long)(size_t)gptr;
  tdm_v4u g0;
  g0.x = 1u;                                             // count=1, user mode
  g0.y = lds_off;                                        // LDS byte address
  g0.z = (unsigned)(ga & 0xFFFFFFFFull);                 // global_addr[31:0]
  g0.w = (unsigned)((ga >> 32) & 0x01FFFFFFull) | (2u << 30); // [56:32]|type=2
  tdm_v8i g1;
  g1[0] = (1 << 16);                                     // data_size = 2 bytes
  g1[1] = (64 << 16);                                    // tensor_dim0 = 64
  g1[2] = (int)(((unsigned)rows_left & 0xFFFFu) << 16);  // tensor_dim1 lo16
  g1[3] = (int)((((unsigned)rows_left >> 16) & 0xFFFFu)  // tensor_dim1 hi16
                | (64u << 16));                          // tile_dim0 = 64
  g1[4] = 64;                                            // tile_dim1=64, dim2=0
  g1[5] = EMBED;                                         // tensor_dim0_stride
  g1[6] = 0;
  g1[7] = 0;
  tdm_v4i z4 = {0, 0, 0, 0};                             // 2D: groups 2/3 unused
  tdm_v8i z8 = {0, 0, 0, 0, 0, 0, 0, 0};
  __builtin_amdgcn_tensor_load_to_lds(g0, g1, z4, z4, z8, 0);
}
#endif

// --------------------------- conversion kernels ----------------------------

__global__ void cvt_f32_bf16(const float* __restrict__ src,
                             __bf16* __restrict__ dst, int n) {
  int i = blockIdx.x * blockDim.x + threadIdx.x;
  if (i < n) dst[i] = f2bf(src[i]);
}

// dst[n*E + k] = src[k*E + n]  (Bt form of a K-major matrix)
__global__ void cvt_T_f32_bf16(const float* __restrict__ src,
                               __bf16* __restrict__ dst) {
  int idx = blockIdx.x * blockDim.x + threadIdx.x;
  if (idx < EMBED * EMBED) {
    int i = idx / EMBED, j = idx % EMBED;
    dst[(size_t)j * EMBED + i] = f2bf(src[idx]);
  }
}

// ------------------------------- WMMA GEMM ---------------------------------
// C[M,N] = A[M,K] @ Bt[N,K]^T.  One wave computes a 32x64 C strip
// (2 M-tiles x 4 N-tiles = 8 WMMA per 32-wide K chunk, 6 fragment loads).
// mode 0: store bf16 row-major C.  mode 1: scatter-store into per-head
//         transposed V layout Vt[(b*512+n)*SEQ + s] (row m = b*SEQ + s).
__global__ void gemm_bt_wmma(const __bf16* __restrict__ A,
                             const __bf16* __restrict__ Bt,
                             __bf16* __restrict__ C,
                             int M, int N, int K, int mode) {
  int lane = threadIdx.x & 31;
  int wid  = blockIdx.x * (blockDim.x >> 5) + (threadIdx.x >> 5);
  int lm = lane & 15;
  int kb = (lane >> 4) * 8;
  int rowbase = (lane >> 4) * 8;

  int ntiles = N >> 6;                 // 64-wide N strips
  int mt = wid / ntiles;
  int n0 = (wid % ntiles) * 64;
  int m0 = mt * 32;
  if (m0 >= M) return;

  v8f acc[2][4] = {};
  for (int kc = 0; kc < K; kc += 32) {
    v16bf a0 = load_frag(A + (size_t)(m0 + lm) * K + kc + kb);
    v16bf a1 = load_frag(A + (size_t)(m0 + 16 + lm) * K + kc + kb);
#pragma unroll
    for (int t = 0; t < 4; ++t) {
      v16bf b = load_frag(Bt + (size_t)(n0 + t * 16 + lm) * K + kc + kb);
      acc[0][t] = wmma_bf16(a0, b, acc[0][t]);
      acc[1][t] = wmma_bf16(a1, b, acc[1][t]);
    }
  }

  if (mode == 0) {
#pragma unroll
    for (int mi = 0; mi < 2; ++mi)
#pragma unroll
      for (int t = 0; t < 4; ++t)
#pragma unroll
        for (int r = 0; r < 8; ++r) {
          int row = m0 + mi * 16 + rowbase + r;
          int col = n0 + t * 16 + lm;
          C[(size_t)row * N + col] = f2bf(acc[mi][t][r]);
        }
  } else {
#pragma unroll
    for (int mi = 0; mi < 2; ++mi)
#pragma unroll
      for (int t = 0; t < 4; ++t)
#pragma unroll
        for (int r = 0; r < 8; ++r) {
          int row = m0 + mi * 16 + rowbase + r;   // row = b*SEQ + s
          int col = n0 + t * 16 + lm;             // col = h*HDIM + d
          int b = row >> 12, s = row & (SEQ - 1);
          C[((size_t)(b * EMBED + col)) * SEQ + s] = f2bf(acc[mi][t][r]);
        }
  }
}

// ---------------------------- flash attention ------------------------------
// Block = 4 waves (128 thr). Each wave owns 16 query rows of one (b,h).
// Online softmax over KV chunks of 64 (4 score tiles, 16 WMMA per chunk);
// P restaged via per-wave LDS tile.  TDM prefetches the next 64x64 K tile
// into a write-only LDS staging buffer (pure L2 warm; read path unchanged).
__global__ void flash_attn_wmma(const __bf16* __restrict__ Q,
                                const __bf16* __restrict__ Km,
                                const __bf16* __restrict__ Vt,
                                float* __restrict__ out) {
  __shared__ __bf16 Pl[4 * 16 * 64];          // 8 KB: one 16x64 tile per wave
#ifdef HAVE_TDM
  __shared__ __bf16 Kstage[64 * 64];          // 8 KB: TDM prefetch target
#endif

  int tid = threadIdx.x;
  int w = tid >> 5, lane = tid & 31;
  int lm = lane & 15;
  int kb = (lane >> 4) * 8;
  int rowbase = (lane >> 4) * 8;
  int woff = w * 16 * 64;

  int bid = blockIdx.x;                       // B*H*(S/64) = 1024 blocks
  int qt = bid & 63;
  int h  = (bid >> 6) & 7;
  int b  = bid >> 9;
  int q0 = qt * 64 + w * 16;

  const __bf16* Qb = Q  + ((size_t)(b * SEQ + q0)) * EMBED + h * HDIM;
  const __bf16* Kb = Km + ((size_t)(b * SEQ))      * EMBED + h * HDIM;
  const __bf16* Vh = Vt + ((size_t)(b * EMBED + h * HDIM)) * SEQ;

#ifdef HAVE_TDM
  unsigned kstage_lds = (unsigned)(size_t)(&Kstage[0]);  // low 32b = LDS addr
#endif

  // Q fragments (16 rows x 64 = two 16x32 operands), kept in registers.
  v16bf qf0 = load_frag(Qb + (size_t)lm * EMBED + kb);
  v16bf qf1 = load_frag(Qb + (size_t)lm * EMBED + 32 + kb);

  float m_i[8], l_i[8];
#pragma unroll
  for (int r = 0; r < 8; ++r) { m_i[r] = -1e30f; l_i[r] = 0.0f; }
  v8f acc[4] = {};

  for (int kv = 0; kv < SEQ; kv += 64) {
#ifdef HAVE_TDM
    // Async TDM prefetch of the NEXT 64x64 K tile (wave 0 only, throttled).
    if (w == 0 && kv + 64 < SEQ) {
      __builtin_amdgcn_s_wait_tensorcnt(4);
      tdm_prefetch_tile(Kb + (size_t)(kv + 64) * EMBED, kstage_lds,
                        SEQ - (kv + 64));
    }
#endif
    __builtin_prefetch(Vh + (size_t)lm * SEQ + kv + 64, 0, 0);

    // scores: S = Q(16x64) @ K_chunk^T as four 16x16 tiles
    v8f s[4] = {};
#pragma unroll
    for (int t4 = 0; t4 < 4; ++t4) {
      const __bf16* Kt = Kb + (size_t)(kv + t4 * 16) * EMBED;
      s[t4] = wmma_bf16(qf0, load_frag(Kt + (size_t)lm * EMBED + kb),      s[t4]);
      s[t4] = wmma_bf16(qf1, load_frag(Kt + (size_t)lm * EMBED + 32 + kb), s[t4]);
      s[t4] = s[t4] * 0.125f;                 // 1/sqrt(64)
    }

    // online softmax: row r lives in VGPR r across 16 lanes of a half-wave
#pragma unroll
    for (int r = 0; r < 8; ++r) {
      float mx = fmaxf(fmaxf(s[0][r], s[1][r]), fmaxf(s[2][r], s[3][r]));
      mx = fmaxf(mx, __shfl_xor(mx, 1, 32));
      mx = fmaxf(mx, __shfl_xor(mx, 2, 32));
      mx = fmaxf(mx, __shfl_xor(mx, 4, 32));
      mx = fmaxf(mx, __shfl_xor(mx, 8, 32));
      float mnew  = fmaxf(m_i[r], mx);
      float alpha = __expf(m_i[r] - mnew);
      float p0 = __expf(s[0][r] - mnew);
      float p1 = __expf(s[1][r] - mnew);
      float p2 = __expf(s[2][r] - mnew);
      float p3 = __expf(s[3][r] - mnew);
      float rs = (p0 + p1) + (p2 + p3);
      rs += __shfl_xor(rs, 1, 32);
      rs += __shfl_xor(rs, 2, 32);
      rs += __shfl_xor(rs, 4, 32);
      rs += __shfl_xor(rs, 8, 32);
      l_i[r] = l_i[r] * alpha + rs;
      m_i[r] = mnew;
      s[0][r] = p0; s[1][r] = p1; s[2][r] = p2; s[3][r] = p3;
      acc[0][r] *= alpha; acc[1][r] *= alpha;
      acc[2][r] *= alpha; acc[3][r] *= alpha;
    }

    // restage P (C-layout) -> LDS -> A-layout operands (same-wave, in-order DS)
#pragma unroll
    for (int t4 = 0; t4 < 4; ++t4)
#pragma unroll
      for (int r = 0; r < 8; ++r)
        Pl[woff + (rowbase + r) * 64 + t4 * 16 + lm] = f2bf(s[t4][r]);
    v16bf pf0 = load_frag(&Pl[woff + lm * 64 + kb]);
    v16bf pf1 = load_frag(&Pl[woff + lm * 64 + 32 + kb]);

    // O += P(16x64) @ V_chunk(64x64); Vt rows give contiguous KV per d
#pragma unroll
    for (int t = 0; t < 4; ++t) {
      v16bf vf0 = load_frag(Vh + (size_t)(t * 16 + lm) * SEQ + kv + kb);
      acc[t] = wmma_bf16(pf0, vf0, acc[t]);
      v16bf vf1 = load_frag(Vh + (size_t)(t * 16 + lm) * SEQ + kv + 32 + kb);
      acc[t] = wmma_bf16(pf1, vf1, acc[t]);
    }
  }

#ifdef HAVE_TDM
  if (w == 0) __builtin_amdgcn_s_wait_tensorcnt(0);   // drain TDM before exit
#endif

  // normalize and store fp32 output [B,S,E]
#pragma unroll
  for (int r = 0; r < 8; ++r) {
    float invl = 1.0f / l_i[r];
    int srow = q0 + rowbase + r;
#pragma unroll
    for (int t = 0; t < 4; ++t)
      out[((size_t)(b * SEQ + srow)) * EMBED + h * HDIM + t * 16 + lm] =
          acc[t][r] * invl;
  }
}

// ------------------------------ host launch --------------------------------

extern "C" void kernel_launch(void* const* d_in, const int* in_sizes, int n_in,
                              void* d_out, int out_size, void* d_ws, size_t ws_size,
                              hipStream_t stream) {
  (void)in_sizes; (void)n_in; (void)out_size;
  const float* rot = (const float*)d_in[0];
  const float* ent = (const float*)d_in[1];
  const float* x   = (const float*)d_in[2];
  const float* wq  = (const float*)d_in[3];
  const float* wk  = (const float*)d_in[4];
  const float* wv  = (const float*)d_in[5];
  float* out = (float*)d_out;

  const int M = BATCH * SEQ;            // 8192
  const int E = EMBED;                  // 512
  const size_t NX = (size_t)M * E;      // 4,194,304
  const size_t NW = (size_t)E * E;      // 262,144

  // bf16 workspace layout (element offsets)
  __bf16* ws = (__bf16*)d_ws;
  size_t off = 0;
  __bf16* Xh  = ws + off; off += NX;
  __bf16* Wqh = ws + off; off += NW;
  __bf16* Wkh = ws + off; off += NW;
  __bf16* Wvh = ws + off; off += NW;
  __bf16* Rt  = ws + off; off += NW;
  __bf16* Et  = ws + off; off += NW;
  __bf16* T0  = ws + off; off += NX;
  __bf16* T1  = ws + off; off += NX;
  __bf16* Qh  = ws + off; off += NX;
  __bf16* Kh  = ws + off; off += NX;
  __bf16* Vt  = ws + off; off += NX;
  if (ws_size < off * sizeof(__bf16)) return;

  // 1) precision conversion (+ transpose for rotation/entangle)
  cvt_f32_bf16<<<(int)((NX + 255) / 256), 256, 0, stream>>>(x,  Xh,  (int)NX);
  cvt_f32_bf16<<<(int)((NW + 255) / 256), 256, 0, stream>>>(wq, Wqh, (int)NW);
  cvt_f32_bf16<<<(int)((NW + 255) / 256), 256, 0, stream>>>(wk, Wkh, (int)NW);
  cvt_f32_bf16<<<(int)((NW + 255) / 256), 256, 0, stream>>>(wv, Wvh, (int)NW);
  cvt_T_f32_bf16<<<(int)((NW + 255) / 256), 256, 0, stream>>>(rot, Rt);
  cvt_T_f32_bf16<<<(int)((NW + 255) / 256), 256, 0, stream>>>(ent, Et);

  // 2) projection GEMMs: (M/32)*(E/64) = 2048 wave-strips -> 512 blocks
  int gemm_blocks = (M / 32) * (E / 64) / 4;
  gemm_bt_wmma<<<gemm_blocks, 128, 0, stream>>>(Xh, Wqh, T0, M, E, E, 0);
  gemm_bt_wmma<<<gemm_blocks, 128, 0, stream>>>(Xh, Wkh, T1, M, E, E, 0);
  gemm_bt_wmma<<<gemm_blocks, 128, 0, stream>>>(Xh, Wvh, Vt, M, E, E, 1); // V^T
  gemm_bt_wmma<<<gemm_blocks, 128, 0, stream>>>(T0, Rt, Qh, M, E, E, 0);
  gemm_bt_wmma<<<gemm_blocks, 128, 0, stream>>>(T1, Et, Kh, M, E, E, 0);

  // 3) flash attention: B*H*(S/64) = 1024 blocks x 128 thr
  flash_attn_wmma<<<BATCH * HEADS * (SEQ / 64), 128, 0, stream>>>(Qh, Kh, Vt, out);
}
